// MultiScaleDynamicVFE_45964740001813
// MI455X (gfx1250) — compile-verified, matrix operand order checked by reading, not run on hardware
//
#include <hip/hip_runtime.h>

// ---------------------------------------------------------------------------
// MultiScaleDynamicVFE for gfx1250 (MI455X).
// WMMA (v_wmma_f32_16x16x32_f16) for the 64->128 PFN layer, fused with
// BN + ReLU + segment-max (uint atomicMax on non-negative floats).
// One wave computes a full 16-row x 128-channel stripe: A fragments are
// loaded once from LDS and reused across all 8 channel tiles.
// ---------------------------------------------------------------------------

typedef __attribute__((ext_vector_type(16))) _Float16 v16h;
typedef __attribute__((ext_vector_type(8)))  _Float16 v8h;
typedef __attribute__((ext_vector_type(8)))  float    v8f;

union ABfrag { v16h v; v8h h[2]; };

#define SXYZc (704 * 800 * 20)   // 11,264,000
#define SYZc  (800 * 20)
#define SZz   20
#define HCAP  (1 << 18)          // hash table capacity (load factor < 0.4)
#define VCAP  (1 << 17)          // 131072 >= max unique voxels (<= N = 100k)
#define KEMPTY 0xFFFFFFFFu

__device__ __forceinline__ unsigned hslot(unsigned k) {
    unsigned h = k * 2654435761u;
    h ^= h >> 15;
    return h & (HCAP - 1);
}

__device__ __forceinline__ int ht_lookup(const unsigned* __restrict__ htk,
                                         const unsigned* __restrict__ htv,
                                         unsigned key) {
    unsigned s = hslot(key);
    for (int it = 0; it < HCAP; ++it) {
        unsigned k2 = htk[s];
        if (k2 == key) return (int)htv[s];
        if (k2 == KEMPTY) return -1;
        s = (s + 1) & (HCAP - 1);
    }
    return -1;
}

// ---------------------------- utility fill ---------------------------------
__global__ void k_fill_u32(unsigned* __restrict__ p, unsigned v, size_t n) {
    size_t i = (size_t)blockIdx.x * blockDim.x + threadIdx.x;
    size_t stride = (size_t)gridDim.x * blockDim.x;
    for (; i < n; i += stride) p[i] = v;
}

// ------------------- level-0 hash build (unique set) -----------------------
__global__ void k_hash_build(const float* __restrict__ pts, int N,
                             unsigned* __restrict__ m0s,
                             unsigned* __restrict__ htk,
                             unsigned* __restrict__ uniq,
                             unsigned* __restrict__ vcnt) {
    int p = blockIdx.x * blockDim.x + threadIdx.x;
    if (p >= N) return;
    int   bi = (int)pts[p * 5 + 0];
    float x = pts[p * 5 + 1], y = pts[p * 5 + 2], z = pts[p * 5 + 3];
    int px = (int)floorf((x - 0.0f)  / 0.1f);
    int py = (int)floorf((y + 40.0f) / 0.1f);
    int pz = (int)floorf((z + 3.0f)  / 0.2f);
    long long m = (long long)bi * SXYZc + (long long)px * SYZc +
                  (long long)py * SZz + pz;
    unsigned key = (unsigned)m;
    m0s[p] = key;
    unsigned s = hslot(key);
    for (int it = 0; it < HCAP; ++it) {
        unsigned old = atomicCAS(&htk[s], KEMPTY, key);
        if (old == KEMPTY) { unsigned idx = atomicAdd(vcnt, 1u); uniq[idx] = key; break; }
        if (old == key) break;
        s = (s + 1) & (HCAP - 1);
    }
}

// ------------------------- bitonic sort step -------------------------------
__global__ void k_bitonic(unsigned* __restrict__ a, int j, int k) {
    int i = blockIdx.x * blockDim.x + threadIdx.x;
    int ixj = i ^ j;
    if (ixj > i && i < VCAP) {
        unsigned ai = a[i], aj = a[ixj];
        bool up = ((i & k) == 0);
        if ((up && ai > aj) || (!up && ai < aj)) { a[i] = aj; a[ixj] = ai; }
    }
}

// -------------------- write sorted ranks into table ------------------------
__global__ void k_rank(const unsigned* __restrict__ uniq,
                       const unsigned* __restrict__ htk,
                       unsigned* __restrict__ htv) {
    int i = blockIdx.x * blockDim.x + threadIdx.x;
    if (i >= VCAP) return;
    unsigned key = uniq[i];
    if (key == KEMPTY) return;
    unsigned s = hslot(key);
    for (int it = 0; it < HCAP; ++it) {
        if (htk[s] == key) { htv[s] = (unsigned)i; return; }
        s = (s + 1) & (HCAP - 1);
    }
}

// ------------- inv0 lookup + per-voxel count/sum (level 0) -----------------
__global__ void k_inv_mean(const float* __restrict__ pts, int N,
                           const unsigned* __restrict__ m0s,
                           const unsigned* __restrict__ htk,
                           const unsigned* __restrict__ htv,
                           int* __restrict__ inv0, int* __restrict__ iout,
                           float* __restrict__ cnt0, float* __restrict__ sum0) {
    int p = blockIdx.x * blockDim.x + threadIdx.x;
    if (p >= N) return;
    int v = ht_lookup(htk, htv, m0s[p]);
    inv0[p] = v;
    iout[p] = v;
    if (v < 0) return;
    atomicAdd(&cnt0[v], 1.0f);
    atomicAdd(&sum0[(size_t)v * 4 + 0], pts[p * 5 + 1]);
    atomicAdd(&sum0[(size_t)v * 4 + 1], pts[p * 5 + 2]);
    atomicAdd(&sum0[(size_t)v * 4 + 2], pts[p * 5 + 3]);
    atomicAdd(&sum0[(size_t)v * 4 + 3], pts[p * 5 + 4]);
}

// --------------- shared PFN layer 0 (10 -> 32, per-row) --------------------
__device__ __forceinline__ void pfn0_row(const float f[10], size_t row, int v,
        const float* __restrict__ pre_g, const float* __restrict__ pre_b,
        const float* __restrict__ pre_rm, const float* __restrict__ pre_rv,
        const float* __restrict__ W0, const float* __restrict__ g0,
        const float* __restrict__ b0, const float* __restrict__ rm0,
        const float* __restrict__ rv0,
        _Float16* __restrict__ harr, float* __restrict__ hmaxarr) {
    float fb[10];
#pragma unroll
    for (int j = 0; j < 10; ++j)
        fb[j] = (f[j] - pre_rm[j]) * (pre_g[j] * rsqrtf(pre_rv[j] + 1e-5f)) + pre_b[j];
#pragma unroll 4
    for (int c = 0; c < 32; ++c) {
        float acc = 0.0f;
#pragma unroll
        for (int j = 0; j < 10; ++j) acc = fmaf(fb[j], W0[j * 32 + c], acc);
        float val = (acc - rm0[c]) * (g0[c] * rsqrtf(rv0[c] + 1e-3f)) + b0[c];
        val = fmaxf(val, 0.0f);
        harr[row * 32 + c] = (_Float16)val;
        atomicMax((unsigned*)&hmaxarr[(size_t)v * 32 + c], __float_as_uint(val));
    }
}

// ----------------------- level-0 PFN layer 0 -------------------------------
__global__ void k_pfn0_l0(const float* __restrict__ pts, int N,
        const int* __restrict__ inv0,
        const float* __restrict__ cnt0, const float* __restrict__ sum0,
        const float* __restrict__ pre_g, const float* __restrict__ pre_b,
        const float* __restrict__ pre_rm, const float* __restrict__ pre_rv,
        const float* __restrict__ W0, const float* __restrict__ g0,
        const float* __restrict__ b0, const float* __restrict__ rm0,
        const float* __restrict__ rv0,
        _Float16* __restrict__ h0, float* __restrict__ hmax0) {
    int p = blockIdx.x * blockDim.x + threadIdx.x;
    if (p >= N) return;
    int v = inv0[p];
    if (v < 0) return;
    float x = pts[p * 5 + 1], y = pts[p * 5 + 2], z = pts[p * 5 + 3];
    float I = pts[p * 5 + 4];
    float rc = 1.0f / cnt0[v];
    float mx = sum0[(size_t)v * 4 + 0] * rc;
    float my = sum0[(size_t)v * 4 + 1] * rc;
    float mz = sum0[(size_t)v * 4 + 2] * rc;
    int px = (int)floorf((x - 0.0f)  / 0.1f);
    int py = (int)floorf((y + 40.0f) / 0.1f);
    int pz = (int)floorf((z + 3.0f)  / 0.2f);
    float f[10] = { x, y, z, I,
                    x - mx, y - my, z - mz,
                    x - ((float)px * 0.1f + 0.05f + 0.0f),
                    y - ((float)py * 0.1f + 0.05f - 40.0f),
                    z - ((float)pz * 0.2f + 0.10f - 3.0f) };
    pfn0_row(f, (size_t)p, v, pre_g, pre_b, pre_rm, pre_rv,
             W0, g0, b0, rm0, rv0, h0, hmax0);
}

// ---------------- level-1 coarse coords (device helper) --------------------
__device__ __forceinline__ void coarse_pc(float x, float y, float z, int o,
                                          int& px, int& py, int& pz) {
    int ix = (o >> 2) & 1, iy = (o >> 1) & 1, iz = o & 1;
    float dx = 0.1f * (0.5f - (float)ix);   // delta = VS0*(0.5 - off)
    float dy = 0.1f * (0.5f - (float)iy);
    float dz = 0.2f * (0.5f - (float)iz);
    px = (int)floorf((x + dx - 0.0f)  / 0.2f) * 2 + ix;
    py = (int)floorf((y + dy + 40.0f) / 0.2f) * 2 + iy;
    pz = (int)floorf((z + dz + 3.0f)  / 0.4f) * 2 + iz;
}

// --------- level-1 candidate classify (isin via hash) + mean sums ----------
__global__ void k_l1_classify(const float* __restrict__ pts, int N,
                              const unsigned* __restrict__ htk,
                              const unsigned* __restrict__ htv,
                              int* __restrict__ r1,
                              float* __restrict__ cnt1, float* __restrict__ sum1) {
    int j = blockIdx.x * blockDim.x + threadIdx.x;
    int NC = 8 * N;
    if (j >= NC) return;
    int o = j / N, p = j - o * N;
    int   bi = (int)pts[p * 5 + 0];
    float x = pts[p * 5 + 1], y = pts[p * 5 + 2], z = pts[p * 5 + 3];
    int px, py, pz;
    coarse_pc(x, y, z, o, px, py, pz);
    long long m = (long long)bi * SXYZc + (long long)px * SYZc +
                  (long long)py * SZz + pz;
    int v = -1;
    if (m >= 0 && m <= (long long)0xFFFFFFFFll)
        v = ht_lookup(htk, htv, (unsigned)m);
    r1[j] = v;
    if (v < 0) return;
    atomicAdd(&cnt1[v], 1.0f);
    atomicAdd(&sum1[(size_t)v * 4 + 0], x);
    atomicAdd(&sum1[(size_t)v * 4 + 1], y);
    atomicAdd(&sum1[(size_t)v * 4 + 2], z);
    atomicAdd(&sum1[(size_t)v * 4 + 3], pts[p * 5 + 4]);
}

// ----------------------- level-1 PFN layer 0 -------------------------------
__global__ void k_pfn0_l1(const float* __restrict__ pts, int N,
        const int* __restrict__ r1,
        const float* __restrict__ cnt1, const float* __restrict__ sum1,
        const float* __restrict__ pre_g, const float* __restrict__ pre_b,
        const float* __restrict__ pre_rm, const float* __restrict__ pre_rv,
        const float* __restrict__ W0, const float* __restrict__ g0,
        const float* __restrict__ b0, const float* __restrict__ rm0,
        const float* __restrict__ rv0,
        _Float16* __restrict__ h1, float* __restrict__ hmax1) {
    int j = blockIdx.x * blockDim.x + threadIdx.x;
    int NC = 8 * N;
    if (j >= NC) return;
    int v = r1[j];
    if (v < 0) return;
    int o = j / N, p = j - o * N;
    float x = pts[p * 5 + 1], y = pts[p * 5 + 2], z = pts[p * 5 + 3];
    float I = pts[p * 5 + 4];
    int px, py, pz;
    coarse_pc(x, y, z, o, px, py, pz);
    float rc = 1.0f / cnt1[v];
    float mx = sum1[(size_t)v * 4 + 0] * rc;
    float my = sum1[(size_t)v * 4 + 1] * rc;
    float mz = sum1[(size_t)v * 4 + 2] * rc;
    float f[10] = { x, y, z, I,
                    x - mx, y - my, z - mz,
                    x - ((float)px * 0.1f + 0.05f + 0.0f),   // f_center uses VS0
                    y - ((float)py * 0.1f + 0.05f - 40.0f),
                    z - ((float)pz * 0.2f + 0.10f - 3.0f) };
    pfn0_row(f, (size_t)j, v, pre_g, pre_b, pre_rm, pre_rv,
             W0, g0, b0, rm0, rv0, h1, hmax1);
}

// ------------- W1 f32 row-major -> f16 column-major (W1t[c*64+k]) ----------
__global__ void k_w1t(const float* __restrict__ W1, _Float16* __restrict__ W1t) {
    int i = blockIdx.x * blockDim.x + threadIdx.x;
    if (i >= 64 * 128) return;
    int k = i >> 7, c = i & 127;
    W1t[(size_t)c * 64 + k] = (_Float16)W1[i];
}

// ---------------- PFN layer 1: WMMA GEMM + BN + ReLU + seg-max -------------
// One wave computes a 16-row x 128-channel stripe: the A tile (16x64 f16,
// K = concat[h(32), hmax(32)]) is staged in LDS once and its two fragments
// are reused across all 8 channel tiles (16 v_wmma_f32_16x16x32_f16 total).
// Voxel ids are staged in LDS so the epilogue reads ds, not global.
__global__ void __launch_bounds__(32)
k_pfn1_gemm(const _Float16* __restrict__ h,
            const float* __restrict__ hmax,
            const int* __restrict__ voxel, int rows,
            const _Float16* __restrict__ W1t,   // [128][64] f16, col-major
            const float* __restrict__ g1,
            const float* __restrict__ b1,
            const float* __restrict__ rm1,
            const float* __restrict__ rv1,
            float* __restrict__ outf, int out_off, int VOUT) {
    __shared__ _Float16 Atile[16 * 64];         // 2 KB
    __shared__ int      Vtile[16];

    const int lane = threadIdx.x;               // 0..31 (wave32)
    const int tile = blockIdx.x;                // row tile

    // ---- cooperative A-tile load: lane -> (row = lane>>1, half = lane&1) ----
    {
        const int r    = lane >> 1;
        const int hsel = lane & 1;              // 0: h[0..31], 1: hmax[0..31]
        const int gr   = tile * 16 + r;
        const int v    = (gr < rows) ? voxel[gr] : -1;
        if (hsel == 0) Vtile[r] = v;
        _Float16* dst = &Atile[r * 64 + hsel * 32];
        if (v >= 0) {
            if (hsel == 0) {
                const v8h* src = (const v8h*)(h + (size_t)gr * 32);
                ((v8h*)dst)[0] = src[0];
                ((v8h*)dst)[1] = src[1];
                ((v8h*)dst)[2] = src[2];
                ((v8h*)dst)[3] = src[3];
            } else {
                const float4* src = (const float4*)(hmax + (size_t)v * 32);
#pragma unroll
                for (int q = 0; q < 8; ++q) {
                    float4 f4 = src[q];
                    dst[q * 4 + 0] = (_Float16)f4.x;
                    dst[q * 4 + 1] = (_Float16)f4.y;
                    dst[q * 4 + 2] = (_Float16)f4.z;
                    dst[q * 4 + 3] = (_Float16)f4.w;
                }
            }
        } else {
            const v8h zz = {};
            ((v8h*)dst)[0] = zz; ((v8h*)dst)[1] = zz;
            ((v8h*)dst)[2] = zz; ((v8h*)dst)[3] = zz;
        }
    }
    __syncthreads();

    // ---- fragment indices (ISA 7.12.2 16-bit layouts) ----
    const int half_sel = lane >> 4;             // 0: lanes 0-15, 1: 16-31
    const int mrow = lane & 15;                 // A row / B column-in-tile
    const int hoff = half_sel * 8;              // K sub-offset per lane half

    // A fragments: loaded once, reused for all 8 channel tiles.
    ABfrag a0, a1;
    a0.h[0] = *(const v8h*)&Atile[mrow * 64 +  0 + hoff];
    a0.h[1] = *(const v8h*)&Atile[mrow * 64 + 16 + hoff];
    a1.h[0] = *(const v8h*)&Atile[mrow * 64 + 32 + hoff];
    a1.h[1] = *(const v8h*)&Atile[mrow * 64 + 48 + hoff];

    // Row voxel ids for this lane's epilogue rows (m = i + half_sel*8).
    int vrow[8];
#pragma unroll
    for (int i = 0; i < 8; ++i) vrow[i] = Vtile[i + (half_sel ? 8 : 0)];

#pragma unroll
    for (int nt = 0; nt < 8; ++nt) {
        const int col = nt * 16 + mrow;         // global output channel
        // B (mirrored layout, lane -> column): contiguous in W1t[col][k]
        const _Float16* bp = W1t + (size_t)col * 64 + hoff;
        ABfrag b0f, b1f;
        b0f.h[0] = *(const v8h*)(bp +  0);
        b0f.h[1] = *(const v8h*)(bp + 16);
        b1f.h[0] = *(const v8h*)(bp + 32);
        b1f.h[1] = *(const v8h*)(bp + 48);

        v8f acc = {};
        acc = __builtin_amdgcn_wmma_f32_16x16x32_f16(
            false, a0.v, false, b0f.v, (short)0, acc, false, false);
        acc = __builtin_amdgcn_wmma_f32_16x16x32_f16(
            false, a1.v, false, b1f.v, (short)0, acc, false, false);

        // BN (per channel == per lane) + ReLU + segment-max scatter
        const float sc = g1[col] * rsqrtf(rv1[col] + 1e-3f);
        const float bs = b1[col] - rm1[col] * sc;
#pragma unroll
        for (int i = 0; i < 8; ++i) {
            const int v2 = vrow[i];
            if (v2 >= 0 && v2 < VOUT) {
                float val = fmaxf(acc[i] * sc + bs, 0.0f);
                atomicMax((unsigned*)&outf[(size_t)v2 * 256 + out_off + col],
                          __float_as_uint(val));
            }
        }
    }
}

// --------------------------- epilogue kernels ------------------------------
__global__ void k_add_emb(float* __restrict__ outf,
                          const float* __restrict__ semb, int VOUT) {
    int i = blockIdx.x * blockDim.x + threadIdx.x;
    if (i >= VOUT * 128) return;
    int v = i >> 7, c = i & 127;
    outf[(size_t)v * 256 + 128 + c] += semb[c];
}

__global__ void k_coords(const unsigned* __restrict__ uniq,
                         int* __restrict__ cout, int VOUT) {
    int i = blockIdx.x * blockDim.x + threadIdx.x;
    if (i >= VOUT) return;
    unsigned m = uniq[i];
    cout[(size_t)i * 4 + 0] = (int)(m / SXYZc);
    unsigned r = m % SXYZc;
    cout[(size_t)i * 4 + 1] = (int)(r / SYZc);
    r %= SYZc;
    cout[(size_t)i * 4 + 2] = (int)(r / SZz);
    cout[(size_t)i * 4 + 3] = (int)(r % SZz);
}

// ---------------------------------------------------------------------------
extern "C" void kernel_launch(void* const* d_in, const int* in_sizes, int n_in,
                              void* d_out, int out_size, void* d_ws, size_t ws_size,
                              hipStream_t stream) {
    const float* pts    = (const float*)d_in[0];
    const float* pre_g  = (const float*)d_in[1];
    const float* pre_b  = (const float*)d_in[2];
    const float* pre_rm = (const float*)d_in[3];
    const float* pre_rv = (const float*)d_in[4];
    const float* W0     = (const float*)d_in[5];
    const float* g0     = (const float*)d_in[6];
    const float* b0     = (const float*)d_in[7];
    const float* rm0    = (const float*)d_in[8];
    const float* rv0    = (const float*)d_in[9];
    const float* W1     = (const float*)d_in[10];
    const float* g1     = (const float*)d_in[11];
    const float* b1     = (const float*)d_in[12];
    const float* rm1    = (const float*)d_in[13];
    const float* rv1    = (const float*)d_in[14];
    const float* semb   = (const float*)d_in[15];   // row 0 used

    const int N  = in_sizes[0] / 5;
    const int NC = 8 * N;
    int VOUT = (out_size - N) / 260;                // out = V*256 + V*4 + N
    if (VOUT < 1) VOUT = 1;

    // ---- workspace carve-up ----
    char* ws = (char*)d_ws;
    size_t off = 0;
    auto alloc = [&](size_t bytes) -> char* {
        off = (off + 255) & ~(size_t)255;
        char* p = ws + off;
        off += bytes;
        return p;
    };
    unsigned* m0s   = (unsigned*)alloc((size_t)N * 4);
    int*      inv0  = (int*)     alloc((size_t)N * 4);
    int*      r1    = (int*)     alloc((size_t)NC * 4);
    unsigned* uniq  = (unsigned*)alloc((size_t)VCAP * 4);
    unsigned* htk   = (unsigned*)alloc((size_t)HCAP * 4);
    unsigned* htv   = (unsigned*)alloc((size_t)HCAP * 4);
    unsigned* vcnt  = (unsigned*)alloc(64);
    float*    cnt0  = (float*)   alloc((size_t)VCAP * 4);
    float*    sum0  = (float*)   alloc((size_t)VCAP * 16);
    float*    cnt1  = (float*)   alloc((size_t)VCAP * 4);
    float*    sum1  = (float*)   alloc((size_t)VCAP * 16);
    float*    hmax0 = (float*)   alloc((size_t)VCAP * 32 * 4);
    float*    hmax1 = (float*)   alloc((size_t)VCAP * 32 * 4);
    _Float16* h0    = (_Float16*)alloc((size_t)N  * 32 * 2);
    _Float16* h1    = (_Float16*)alloc((size_t)NC * 32 * 2);
    _Float16* W1t   = (_Float16*)alloc((size_t)64 * 128 * 2);

    float* fout = (float*)d_out;
    int*   cout = (int*)d_out + (size_t)VOUT * 256;
    int*   iout = cout + (size_t)VOUT * 4;

    const int TB = 256;
    // ---- init ----
    k_fill_u32<<<2048, TB, 0, stream>>>(htk,  KEMPTY, (size_t)HCAP);
    k_fill_u32<<<2048, TB, 0, stream>>>(uniq, KEMPTY, (size_t)VCAP);
    k_fill_u32<<<1,    64, 0, stream>>>(vcnt, 0u, 16);
    k_fill_u32<<<2048, TB, 0, stream>>>((unsigned*)cnt0, 0u, (size_t)VCAP);
    k_fill_u32<<<2048, TB, 0, stream>>>((unsigned*)sum0, 0u, (size_t)VCAP * 4);
    k_fill_u32<<<2048, TB, 0, stream>>>((unsigned*)cnt1, 0u, (size_t)VCAP);
    k_fill_u32<<<2048, TB, 0, stream>>>((unsigned*)sum1, 0u, (size_t)VCAP * 4);
    k_fill_u32<<<4096, TB, 0, stream>>>((unsigned*)hmax0, 0u, (size_t)VCAP * 32);
    k_fill_u32<<<4096, TB, 0, stream>>>((unsigned*)hmax1, 0u, (size_t)VCAP * 32);
    k_fill_u32<<<4096, TB, 0, stream>>>((unsigned*)fout, 0u, (size_t)VOUT * 256);

    // ---- level-0 voxel hashing + sorted unique (bitonic) ----
    k_hash_build<<<(N + TB - 1) / TB, TB, 0, stream>>>(pts, N, m0s, htk, uniq, vcnt);
    for (int k = 2; k <= VCAP; k <<= 1)
        for (int j = k >> 1; j > 0; j >>= 1)
            k_bitonic<<<VCAP / TB, TB, 0, stream>>>(uniq, j, k);
    k_rank<<<VCAP / TB, TB, 0, stream>>>(uniq, htk, htv);
    k_inv_mean<<<(N + TB - 1) / TB, TB, 0, stream>>>(pts, N, m0s, htk, htv,
                                                     inv0, iout, cnt0, sum0);

    // ---- PFN stage 0 (level 0) ----
    k_pfn0_l0<<<(N + TB - 1) / TB, TB, 0, stream>>>(pts, N, inv0, cnt0, sum0,
            pre_g, pre_b, pre_rm, pre_rv, W0, g0, b0, rm0, rv0, h0, hmax0);

    // ---- PFN stage 1 (WMMA) level 0 ----
    k_w1t<<<(64 * 128 + TB - 1) / TB, TB, 0, stream>>>(W1, W1t);
    k_pfn1_gemm<<<(N + 15) / 16, 32, 0, stream>>>(h0, hmax0, inv0, N, W1t,
                                                  g1, b1, rm1, rv1, fout, 0, VOUT);

    // ---- level 1 ----
    k_l1_classify<<<(NC + TB - 1) / TB, TB, 0, stream>>>(pts, N, htk, htv,
                                                         r1, cnt1, sum1);
    k_pfn0_l1<<<(NC + TB - 1) / TB, TB, 0, stream>>>(pts, N, r1, cnt1, sum1,
            pre_g, pre_b, pre_rm, pre_rv, W0, g0, b0, rm0, rv0, h1, hmax1);
    k_pfn1_gemm<<<(NC + 15) / 16, 32, 0, stream>>>(h1, hmax1, r1, NC, W1t,
                                                   g1, b1, rm1, rv1, fout, 128, VOUT);

    // ---- epilogue ----
    k_add_emb<<<(VOUT * 128 + TB - 1) / TB, TB, 0, stream>>>(fout, semb, VOUT);
    k_coords<<<(VOUT + TB - 1) / TB, TB, 0, stream>>>(uniq, cout, VOUT);
}
